// AttentionChannelPonder_47407849013402
// MI455X (gfx1250) — matrix operand order repaired and checked by reading, not compiled, not run on hardware
//
#include <hip/hip_runtime.h>

// ---------------- types ----------------
typedef __bf16 bf16_t;
typedef __attribute__((ext_vector_type(16))) __bf16 v16bf;
typedef __attribute__((ext_vector_type(8)))  float  v8f;
typedef int v4i_vec __attribute__((vector_size(16)));   // matches builtin proto

#define B_DIM 8
#define C_DIM 8
#define T_DIM 2048
#define H_DIM 256
#define NH_DIM 4
#define F_BINS 257
#define WIN_DIM 512
#define ROWS (B_DIM * T_DIM)          // 16384
#define ACT_EPS 1e-4f
#define LDSS 40                        // padded LDS row stride (bf16 elems)

#ifndef __has_builtin
#define __has_builtin(x) 0
#endif

#if defined(__AMDGCN__) && __has_builtin(__builtin_amdgcn_global_load_async_to_lds_b128)
#define HAVE_ASYNC_LDS 1
#else
#define HAVE_ASYNC_LDS 0
#endif

#define AS1 __attribute__((address_space(1)))
#define AS3 __attribute__((address_space(3)))

// 16-byte global -> LDS copy; async DMA (ASYNCcnt) when available.
__device__ __forceinline__ void async_ld_b128(void* lds, const void* g)
{
#if HAVE_ASYNC_LDS
    __builtin_amdgcn_global_load_async_to_lds_b128(
        (AS1 v4i_vec*)(AS1 void*)(void*)g,
        (AS3 v4i_vec*)(AS3 void*)lds,
        0, 0);
#else
    *(uint4*)lds = *(const uint4*)g;
#endif
}

__device__ __forceinline__ void wait_async_lds()
{
#if HAVE_ASYNC_LDS
#if __has_builtin(__builtin_amdgcn_s_wait_asynccnt)
    __builtin_amdgcn_s_wait_asynccnt(0);
#else
    asm volatile("s_wait_asynccnt 0" ::: "memory");
#endif
#endif
}

// =====================================================================
// WMMA GEMM:  C[M,N] = act(A[M,K] @ B[K,N] + bias)
// Bt[N][ldb] is pre-transposed bf16, ldb = K rounded up to 32 (zero pad).
// A: bf16 (optional per-row gather over lin1 [B,C,T,H]) or fp32.
// Block: 128 thr = 4 waves; tile 128(M) x 64(N); K-step 32; LDS double-buffered.
// Wave w: rows [m0+32w, +32) x 64 cols -> 2 A-frags x 4 B-frags -> 8 WMMA/step.
// bf16 tiles staged with GLOBAL_LOAD_ASYNC_TO_LDS_B128 (ASYNCcnt DMA).
// =====================================================================
template<bool A_F32, bool RELU, bool OUT_BF16>
__global__ __launch_bounds__(128)
void gemm_wmma(const void* __restrict__ Av, const bf16_t* __restrict__ Bt,
               const float* __restrict__ bias, void* __restrict__ Cv,
               int M, int N, int K, int lda, int ldb, int ldc,
               int gcc, int gch0)
{
    __shared__ bf16_t As[2][128][LDSS];
    __shared__ bf16_t Bs[2][64][LDSS];

    const int tid  = threadIdx.x;
    const int wave = tid >> 5;
    const int lane = tid & 31;
    const int m0   = blockIdx.y * 128;
    const int n0   = blockIdx.x * 64;

    // physical A row for this thread's staging row (tid = row in tile)
    long arow;
    {
        int r = m0 + tid;
        if (gcc > 0) {
            int per = gcc * T_DIM;
            int b   = r / per;
            int rem = r - b * per;
            int ch  = rem / T_DIM;
            int t   = rem - ch * T_DIM;
            arow = (long)(b * C_DIM + gch0 + ch) * T_DIM + t;
        } else {
            arow = r;
        }
    }
    const float*  Af = (const float*)Av;
    const bf16_t* Ab = (const bf16_t*)Av;

    auto stageA = [&](int buf, int k0) {
        const int row = tid;
        if (A_F32) {
            #pragma unroll
            for (int h = 0; h < 2; ++h) {
                int kc = k0 + h * 16;
                union { bf16_t e[16]; uint4 q[2]; } tmp;
                if (kc + 16 <= K) {
                    const float* src = Af + arow * (long)lda + kc;
                    #pragma unroll
                    for (int i = 0; i < 16; ++i) tmp.e[i] = (bf16_t)src[i];
                } else {
                    #pragma unroll
                    for (int i = 0; i < 16; ++i) {
                        int c = kc + i;
                        tmp.e[i] = (bf16_t)((c < K) ? Af[arow * (long)lda + c] : 0.f);
                    }
                }
                uint4* dst = (uint4*)&As[buf][row][h * 16];
                dst[0] = tmp.q[0]; dst[1] = tmp.q[1];
            }
        } else {
            // bf16 A: K is a multiple of 32 for all bf16-A GEMMs -> no guards
            const bf16_t* src = Ab + arow * (long)lda + k0;
            #pragma unroll
            for (int c = 0; c < 4; ++c)
                async_ld_b128(&As[buf][row][c * 8], src + c * 8);
        }
    };
    auto stageB = [&](int buf, int k0) {
        const int nr = tid >> 1, hf = tid & 1;
        int n = n0 + nr;
        if (n < N) {   // ldb zero-padded to >= any staged chunk
            const bf16_t* src = Bt + (long)n * ldb + k0 + hf * 16;
            async_ld_b128(&Bs[buf][nr][hf * 16],     src);
            async_ld_b128(&Bs[buf][nr][hf * 16 + 8], src + 8);
        }
    };

    v8f acc[2][4] = {};

    const int lane15 = lane & 15;
    const int klA    = (lane >> 4) * 8;    // unused; A fragments also use kbB
    const int kbB    = (lane >> 4) * 16;   // B fragment K sub-offset
    const int nSteps = (K + 31) / 32;

    stageA(0, 0);
    stageB(0, 0);

    for (int s = 0; s < nSteps; ++s) {
        const int buf = s & 1;
        wait_async_lds();       // own async writes to `buf` done
        __syncthreads();        // everyone's writes to `buf` visible
        if (s + 1 < nSteps) {   // stage next tile into other buffer
            stageA(buf ^ 1, (s + 1) * 32);
            stageB(buf ^ 1, (s + 1) * 32);
        }
        // ---- preload all fragments, then back-to-back WMMAs ----
        union { v16bf v; uint4 q[2]; } afr[2];
        #pragma unroll
        for (int mi = 0; mi < 2; ++mi) {
            const bf16_t* ap = &As[buf][wave * 32 + mi * 16 + lane15][0];
            afr[mi].q[0] = *(const uint4*)(ap + kbB);
            afr[mi].q[1] = *(const uint4*)(ap + kbB + 8);
        }
        union { v16bf v; uint4 q[2]; } bfr[4];
        #pragma unroll
        for (int nt = 0; nt < 4; ++nt) {
            const bf16_t* bp = &Bs[buf][nt * 16 + lane15][kbB];
            bfr[nt].q[0] = *(const uint4*)(bp);
            bfr[nt].q[1] = *(const uint4*)(bp + 8);
        }
        #pragma unroll
        for (int mi = 0; mi < 2; ++mi)
            #pragma unroll
            for (int nt = 0; nt < 4; ++nt)
                acc[mi][nt] = __builtin_amdgcn_wmma_f32_16x16x32_bf16(
                    false, afr[mi].v, false, bfr[nt].v, (short)0,
                    acc[mi][nt], false, false);
        __syncthreads();        // done reading `buf` before it is restaged
    }

    // ---- epilogue: bias (+relu), store fp32 or bf16 ----
    #pragma unroll
    for (int mi = 0; mi < 2; ++mi) {
        int mb = m0 + wave * 32 + mi * 16 + ((lane >> 4) * 8);
        #pragma unroll
        for (int nt = 0; nt < 4; ++nt) {
            int n = n0 + nt * 16 + lane15;
            if (n >= N) continue;
            float bv = bias[n];
            #pragma unroll
            for (int i = 0; i < 8; ++i) {
                int m = mb + i;
                if (m >= M) continue;
                float v = acc[mi][nt][i] + bv;
                if (RELU) v = v > 0.f ? v : 0.f;
                if (OUT_BF16) ((bf16_t*)Cv)[(long)m * ldc + n] = (bf16_t)v;
                else          ((float*) Cv)[(long)m * ldc + n] = v;
            }
        }
    }
}

// =====================================================================
// Weight prep: dst[N][Kp] = bf16(src[K][N]) transposed, zero-padded cols.
// =====================================================================
__global__ void wprep_kernel(const float* __restrict__ src, bf16_t* __restrict__ dst,
                             int K, int N, int Kp)
{
    int idx = blockIdx.x * blockDim.x + threadIdx.x;
    if (idx >= N * Kp) return;
    int n = idx / Kp, k = idx - n * Kp;
    float v = (k < K) ? src[(long)k * N + n] : 0.f;
    dst[(long)n * Kp + k] = (bf16_t)v;
}

// =====================================================================
// Channel attention per (b,t,head): softmax over cc channels (wave/item)
// K/V cached as [ch][b][t][H]  (channel-major, appended incrementally)
// =====================================================================
__global__ __launch_bounds__(128)
void attn_kernel(const bf16_t* __restrict__ Q, const bf16_t* __restrict__ Kc,
                 const bf16_t* __restrict__ Vc, bf16_t* __restrict__ ctx, int cc)
{
    int wave = threadIdx.x >> 5, lane = threadIdx.x & 31;
    int gw   = blockIdx.x * 4 + wave;       // row*NH + head
    int row  = gw >> 2, head = gw & 3;
    int b    = row >> 11, t = row & (T_DIM - 1);

    long qoff = (long)row * H_DIM + head * 64 + lane * 2;
    float q0 = (float)Q[qoff], q1 = (float)Q[qoff + 1];

    float sc[C_DIM];
    #pragma unroll
    for (int ch = 0; ch < C_DIM; ++ch) {
        if (ch < cc) {
            long koff = ((long)(ch * B_DIM + b) * T_DIM + t) * H_DIM + head * 64 + lane * 2;
            float p = q0 * (float)Kc[koff] + q1 * (float)Kc[koff + 1];
            #pragma unroll
            for (int off = 16; off >= 1; off >>= 1) p += __shfl_xor(p, off, 32);
            sc[ch] = p * 0.125f;              // / sqrt(DH=64)
        }
    }
    float mx = sc[0];
    #pragma unroll
    for (int ch = 1; ch < C_DIM; ++ch) if (ch < cc) mx = fmaxf(mx, sc[ch]);
    float sum = 0.f;
    #pragma unroll
    for (int ch = 0; ch < C_DIM; ++ch) if (ch < cc) { sc[ch] = __expf(sc[ch] - mx); sum += sc[ch]; }
    float inv = 1.f / sum;
    float c0 = 0.f, c1 = 0.f;
    #pragma unroll
    for (int ch = 0; ch < C_DIM; ++ch) {
        if (ch < cc) {
            long voff = ((long)(ch * B_DIM + b) * T_DIM + t) * H_DIM + head * 64 + lane * 2;
            float a = sc[ch] * inv;
            c0 += a * (float)Vc[voff];
            c1 += a * (float)Vc[voff + 1];
        }
    }
    ctx[qoff]     = (bf16_t)c0;
    ctx[qoff + 1] = (bf16_t)c1;
}

// =====================================================================
// LayerNorm(pre + resid) row-wise (H=256), wave per row.
// resid optionally gathered from lin1 channel layout.
// =====================================================================
__global__ __launch_bounds__(128)
void ln_kernel(const float* __restrict__ pre, const bf16_t* __restrict__ resid,
               int rgcc, int rch0, const float* __restrict__ g,
               const float* __restrict__ bta, bf16_t* __restrict__ out)
{
    int wave = threadIdx.x >> 5, lane = threadIdx.x & 31;
    int row  = blockIdx.x * 4 + wave;
    long rrow = row;
    if (rgcc > 0) {
        int b = row >> 11, t = row & (T_DIM - 1);
        rrow = (long)(b * C_DIM + rch0) * T_DIM + t;
    }
    float x[8]; float s = 0.f;
    #pragma unroll
    for (int i = 0; i < 8; ++i) {
        int d = lane + 32 * i;
        float v = pre[(long)row * H_DIM + d] + (float)resid[rrow * H_DIM + d];
        x[i] = v; s += v;
    }
    #pragma unroll
    for (int off = 16; off >= 1; off >>= 1) s += __shfl_xor(s, off, 32);
    float mean = s * (1.f / 256.f);
    float ss = 0.f;
    #pragma unroll
    for (int i = 0; i < 8; ++i) { float d0 = x[i] - mean; ss += d0 * d0; }
    #pragma unroll
    for (int off = 16; off >= 1; off >>= 1) ss += __shfl_xor(ss, off, 32);
    float rstd = rsqrtf(ss * (1.f / 256.f) + 1e-5f);
    #pragma unroll
    for (int i = 0; i < 8; ++i) {
        int d = lane + 32 * i;
        out[(long)row * H_DIM + d] = (bf16_t)((x[i] - mean) * rstd * g[d] + bta[d]);
    }
}

// =====================================================================
// Halting score tail: cur = sigmoid(((s1@Ws2)+bs2 - ps_a)/(ps_b+1e-8))
// =====================================================================
__global__ __launch_bounds__(128)
void score_kernel(const bf16_t* __restrict__ s1, const float* __restrict__ Ws2,
                  const float* __restrict__ bs2, const float* __restrict__ ps_a,
                  const float* __restrict__ ps_b, float* __restrict__ cur)
{
    int wave = threadIdx.x >> 5, lane = threadIdx.x & 31;
    int row  = blockIdx.x * 4 + wave;
    float s = 0.f;
    #pragma unroll
    for (int i = 0; i < 4; ++i) {
        int d = lane + 32 * i;
        s += (float)s1[(long)row * 128 + d] * Ws2[d];
    }
    #pragma unroll
    for (int off = 16; off >= 1; off >>= 1) s += __shfl_xor(s, off, 32);
    if (lane == 0) {
        float v = (s + bs2[0] - ps_a[0]) / (ps_b[0] + 1e-8f);
        cur[row] = 1.f / (1.f + __expf(-v));
    }
}

// =====================================================================
// ACT ponder-state update; wave per row; gate is a device scalar.
// =====================================================================
__global__ __launch_bounds__(128)
void act_kernel(const bf16_t* __restrict__ hh, const float* __restrict__ cur,
                float* __restrict__ out, float* __restrict__ cum,
                float* __restrict__ rem, float* __restrict__ still,
                float* __restrict__ pc, const float* __restrict__ gate,
                float* __restrict__ ssum, int isLast)
{
    int wave = threadIdx.x >> 5, lane = threadIdx.x & 31;
    int r    = blockIdx.x * 4 + wave;

    float gv   = gate[0];
    int   live = gv > 0.5f;
    float st = still[r], cm = cum[r], rm = rem[r], p = pc[r];
    float cr = isLast ? 1.0f : cur[r];

    float cum_n   = cm + cr * st;
    float pc_n    = p + st;
    float still_n = (cum_n < 1.0f - ACT_EPS) ? 1.0f : 0.0f;
    float coef    = cr * still_n + rm * (1.0f - still_n);
    float rem_n   = rm - cr * still_n;
    pc_n         += rem_n * (1.0f - still_n);

    if (live) {
        #pragma unroll
        for (int i = 0; i < 8; ++i) {
            int d = lane + 32 * i;
            out[(long)r * H_DIM + d] += coef * (float)hh[(long)r * H_DIM + d];
        }
    }
    if (lane == 0) {
        float stillFinal = live ? still_n : st;
        if (live) { cum[r] = cum_n; rem[r] = rem_n; still[r] = still_n; pc[r] = pc_n; }
        atomicAdd(ssum, stillFinal);
    }
}

__global__ void gateupd_kernel(float* gate, float* ssum)
{
    float s = *ssum;
    gate[0] = gate[0] * ((s >= ACT_EPS) ? 1.0f : 0.0f);
    *ssum = 0.f;
}

__global__ void init_kernel(float* out, float* cum, float* rem, float* still,
                            float* pc, float* gate, float* ssum)
{
    long idx = (long)blockIdx.x * blockDim.x + threadIdx.x;
    if (idx < (long)ROWS * H_DIM) out[idx] = 0.f;
    if (idx < ROWS) { cum[idx] = 0.f; pc[idx] = 0.f; rem[idx] = 1.f; still[idx] = 1.f; }
    if (idx == 0) { gate[0] = 1.f; ssum[0] = 0.f; }
}

__global__ void pccopy_kernel(const float* __restrict__ pc, float* __restrict__ dst)
{
    int idx = blockIdx.x * blockDim.x + threadIdx.x;
    if (idx < ROWS) dst[idx] = pc[idx];
}

// =====================================================================
// Host orchestration
// =====================================================================
template<bool AF32, bool RELU, bool OBF>
static void launch_gemm(hipStream_t stream, const void* A, const bf16_t* Bt,
                        const float* bias, void* Cp, int M, int N, int K,
                        int lda, int ldb, int ldc, int gcc, int gch0)
{
    dim3 grid((N + 63) / 64, (M + 127) / 128), block(128);
    gemm_wmma<AF32, RELU, OBF><<<grid, block, 0, stream>>>(
        A, Bt, bias, Cp, M, N, K, lda, ldb, ldc, gcc, gch0);
}

static inline size_t align256(size_t x) { return (x + 255) & ~(size_t)255; }

extern "C" void kernel_launch(void* const* d_in, const int* in_sizes, int n_in,
                              void* d_out, int out_size, void* d_ws, size_t ws_size,
                              hipStream_t stream)
{
    (void)in_sizes; (void)n_in; (void)out_size; (void)ws_size;
    const float* x     = (const float*)d_in[0];
    const float* W_in  = (const float*)d_in[1];
    const float* b_in  = (const float*)d_in[2];
    const float* W_out = (const float*)d_in[3];
    const float* b_out = (const float*)d_in[4];
    const float* Wq    = (const float*)d_in[5];
    const float* bq    = (const float*)d_in[6];
    const float* Wk    = (const float*)d_in[7];
    const float* bk    = (const float*)d_in[8];
    const float* Wv    = (const float*)d_in[9];
    const float* bv    = (const float*)d_in[10];
    const float* Wo    = (const float*)d_in[11];
    const float* bo    = (const float*)d_in[12];
    const float* ln1g  = (const float*)d_in[13];
    const float* ln1b  = (const float*)d_in[14];
    const float* Wf1   = (const float*)d_in[15];
    const float* bf1   = (const float*)d_in[16];
    const float* Wf2   = (const float*)d_in[17];
    const float* bf2   = (const float*)d_in[18];
    const float* ln2g  = (const float*)d_in[19];
    const float* ln2b  = (const float*)d_in[20];
    const float* Ws1   = (const float*)d_in[21];
    const float* bs1   = (const float*)d_in[22];
    const float* Ws2   = (const float*)d_in[23];
    const float* bs2   = (const float*)d_in[24];
    const float* ps_a  = (const float*)d_in[25];
    const float* ps_b  = (const float*)d_in[26];

    // ---- workspace carve ----
    char* ws = (char*)d_ws;
    size_t off = 0;
    auto carve = [&](size_t bytes) -> void* {
        void* p = ws + off; off = align256(off + bytes); return p;
    };
    bf16_t* WtIn  = (bf16_t*)carve((size_t)256 * 288 * 2);   // [256][288], K=257 padded to 32-mult
    bf16_t* WtQ   = (bf16_t*)carve((size_t)256 * 256 * 2);
    bf16_t* WtK   = (bf16_t*)carve((size_t)256 * 256 * 2);
    bf16_t* WtV   = (bf16_t*)carve((size_t)256 * 256 * 2);
    bf16_t* WtO   = (bf16_t*)carve((size_t)256 * 256 * 2);
    bf16_t* WtF1  = (bf16_t*)carve((size_t)512 * 256 * 2);
    bf16_t* WtF2  = (bf16_t*)carve((size_t)256 * 512 * 2);
    bf16_t* WtOut = (bf16_t*)carve((size_t)257 * 256 * 2);
    bf16_t* WtS1  = (bf16_t*)carve((size_t)128 * 256 * 2);
    bf16_t* lin1  = (bf16_t*)carve((size_t)B_DIM * C_DIM * T_DIM * H_DIM * 2);
    bf16_t* Qb    = (bf16_t*)carve((size_t)ROWS * H_DIM * 2);
    bf16_t* Kb    = (bf16_t*)carve((size_t)C_DIM * ROWS * H_DIM * 2);
    bf16_t* Vb    = (bf16_t*)carve((size_t)C_DIM * ROWS * H_DIM * 2);
    bf16_t* ctx   = (bf16_t*)carve((size_t)ROWS * H_DIM * 2);
    float*  pre   = (float*) carve((size_t)ROWS * H_DIM * 4);
    bf16_t* h1    = (bf16_t*)carve((size_t)ROWS * H_DIM * 2);
    bf16_t* f1    = (bf16_t*)carve((size_t)ROWS * WIN_DIM * 2);
    bf16_t* hh    = (bf16_t*)carve((size_t)ROWS * H_DIM * 2);
    bf16_t* s1b   = (bf16_t*)carve((size_t)ROWS * 128 * 2);
    float*  cur   = (float*) carve((size_t)ROWS * 4);
    float*  outA  = (float*) carve((size_t)ROWS * H_DIM * 4);
    float*  cumB  = (float*) carve((size_t)ROWS * 4);
    float*  remB  = (float*) carve((size_t)ROWS * 4);
    float*  stB   = (float*) carve((size_t)ROWS * 4);
    float*  pcB   = (float*) carve((size_t)ROWS * 4);
    float*  gate  = (float*) carve(4);
    float*  ssum  = (float*) carve(4);

    // ---- weight transpose/convert to bf16 ----
    auto WP = [&](const float* s, bf16_t* d, int K, int N, int Kp) {
        int tot = N * Kp;
        wprep_kernel<<<(tot + 255) / 256, 256, 0, stream>>>(s, d, K, N, Kp);
    };
    WP(W_in,  WtIn,  257, 256, 288);
    WP(Wq,    WtQ,   256, 256, 256);
    WP(Wk,    WtK,   256, 256, 256);
    WP(Wv,    WtV,   256, 256, 256);
    WP(Wo,    WtO,   256, 256, 256);
    WP(Wf1,   WtF1,  256, 512, 256);
    WP(Wf2,   WtF2,  512, 256, 512);
    WP(W_out, WtOut, 256, 257, 256);
    WP(Ws1,   WtS1,  256, 128, 256);

    init_kernel<<<(ROWS * H_DIM) / 256, 256, 0, stream>>>(outA, cumB, remB, stB, pcB, gate, ssum);

    // lin1 = relu(x @ W_in + b_in)  [131072 x 256], K=257, A fp32
    launch_gemm<true, true, true>(stream, x, WtIn, b_in, lin1,
                                  B_DIM * C_DIM * T_DIM, 256, 257, 257, 288, 256, 0, 0);

    for (int ci = 0; ci < C_DIM; ++ci) {
        const int cc = ci + 1;
        // Q for channel ci; K/V appended only for the NEW channel (loop-invariant weights)
        launch_gemm<false, false, true>(stream, lin1, WtQ, bq, Qb,
                                        ROWS, 256, 256, 256, 256, 256, 1, ci);
        launch_gemm<false, false, true>(stream, lin1, WtK, bk,
                                        Kb + (size_t)ci * ROWS * H_DIM,
                                        ROWS, 256, 256, 256, 256, 256, 1, ci);
        launch_gemm<false, false, true>(stream, lin1, WtV, bv,
                                        Vb + (size_t)ci * ROWS * H_DIM,
                                        ROWS, 256, 256, 256, 256, 256, 1, ci);
        attn_kernel<<<ROWS * NH_DIM / 4, 128, 0, stream>>>(Qb, Kb, Vb, ctx, cc);

        // h1 = LN(q + ctx@Wo + bo)
        launch_gemm<false, false, false>(stream, ctx, WtO, bo, pre,
                                         ROWS, 256, 256, 256, 256, 256, 0, 0);
        ln_kernel<<<ROWS / 4, 128, 0, stream>>>(pre, lin1, 1, ci, ln1g, ln1b, h1);

        // f = relu(h1@Wf1+bf1)@Wf2+bf2 ; hh = LN(h1 + f)
        launch_gemm<false, true, true>(stream, h1, WtF1, bf1, f1,
                                       ROWS, 512, 256, 256, 256, 512, 0, 0);
        launch_gemm<false, false, false>(stream, f1, WtF2, bf2, pre,
                                         ROWS, 256, 512, 512, 512, 256, 0, 0);
        ln_kernel<<<ROWS / 4, 128, 0, stream>>>(pre, h1, 0, 0, ln2g, ln2b, hh);

        if (ci < C_DIM - 1) {
            launch_gemm<false, true, true>(stream, hh, WtS1, bs1, s1b,
                                           ROWS, 128, 256, 256, 256, 128, 0, 0);
            score_kernel<<<ROWS / 4, 128, 0, stream>>>(s1b, Ws2, bs2, ps_a, ps_b, cur);
        }
        act_kernel<<<ROWS / 4, 128, 0, stream>>>(hh, cur, outA, cumB, remB, stB, pcB,
                                                 gate, ssum, (ci == C_DIM - 1) ? 1 : 0);
        gateupd_kernel<<<1, 1, 0, stream>>>(gate, ssum);
    }

    // encoder_out = relu(out @ W_out + b_out) -> d_out fp32 [16384 x 257]
    launch_gemm<true, true, false>(stream, outA, WtOut, b_out, d_out,
                                   ROWS, F_BINS, 256, 256, 256, F_BINS, 0, 0);
    pccopy_kernel<<<(ROWS + 255) / 256, 256, 0, stream>>>(pcB, (float*)d_out + (size_t)ROWS * F_BINS);
}
